// Attention_23725399343323
// MI455X (gfx1250) — compile-verified, matrix operand order checked by reading, not run on hardware
//
#include <hip/hip_runtime.h>

// ---------------- problem constants ----------------
#define BATCH 16
#define NTOK  4096      // H*W = 64*64
#define CDIM  128
#define HEADS 2
#define DH    64
#define SRF   4
#define NSR   256       // (64/4)*(64/4)
#define KPATCH 2048     // C*SR*SR
#define WW    64

typedef __attribute__((ext_vector_type(16))) __bf16 v16bf;
typedef __attribute__((ext_vector_type(8)))  float  v8f;

__device__ __forceinline__ unsigned short f2bf(float f) {
    unsigned int u = __float_as_uint(f);
    unsigned int r = u + 0x7FFFu + ((u >> 16) & 1u);   // round-to-nearest-even
    return (unsigned short)(r >> 16);
}

// ---------------- fp32 -> bf16 convert ----------------
__global__ __launch_bounds__(256) void cvt_f32_bf16(const float* __restrict__ src,
                                                    unsigned short* __restrict__ dst,
                                                    int n) {
    int i = blockIdx.x * blockDim.x + threadIdx.x;
    int stride = gridDim.x * blockDim.x;
    for (; i < n; i += stride) dst[i] = f2bf(src[i]);
}

// ---------------- Wsr permute+convert: [co][ci][kh][kw] -> [co][(kh*4+kw)*128 + ci] bf16 ----------------
__global__ __launch_bounds__(256) void cvt_wsr_perm(const float* __restrict__ src,
                                                    unsigned short* __restrict__ dst) {
    int i = blockIdx.x * blockDim.x + threadIdx.x;
    const int total = CDIM * KPATCH;
    if (i >= total) return;
    int co  = i >> 11;
    int r   = i & 2047;
    int pix = r >> 7;            // kh*4+kw
    int ci  = r & 127;
    int kh  = pix >> 2;
    int kw  = pix & 3;
    dst[i] = f2bf(src[((co * CDIM + ci) * SRF + kh) * SRF + kw]);
}

// ---------------- generic bf16 WMMA GEMM: D[m,n] = (sum_k A[m,k]*Bw[n,k] + bias[n]) * outScale ----------------
// One 16x16 output tile per wave32; K multiple of 32; M,N multiples of 16.
__global__ __launch_bounds__(256) void gemm_bf16_wmma(
    const unsigned short* __restrict__ A, long long lda,
    const unsigned short* __restrict__ Bw, long long ldb,
    float* __restrict__ outF, unsigned short* __restrict__ outB, long long ldo,
    const float* __restrict__ bias,
    int M, int N, int K, float outScale)
{
    const int tilesN = N >> 4;
    const int tilesM = M >> 4;
    const int wave = blockIdx.x * (blockDim.x >> 5) + (threadIdx.x >> 5);
    if (wave >= tilesM * tilesN) return;            // wave-uniform exit
    const int tm = wave / tilesN;
    const int tn = wave % tilesN;

    const int lane  = threadIdx.x & 31;
    const int lr    = lane & 15;                    // A-row / B-col / C-col within tile
    const int khalf = lane >> 4;                    // K chunk split per 16-bit A/B layout

    const unsigned short* aRow = A  + (long long)(tm * 16 + lr) * lda;
    const unsigned short* bRow = Bw + (long long)(tn * 16 + lr) * ldb;

    v8f acc = {};
    for (int k = 0; k < K; k += 32) {
        union { v16bf v; uint4 q[2]; } af, bf;
        const unsigned short* ap = aRow + k + khalf * 8;
        af.q[0] = *(const uint4*)(ap);
        af.q[1] = *(const uint4*)(ap + 16);
        const unsigned short* bp = bRow + k + khalf * 8;
        bf.q[0] = *(const uint4*)(bp);
        bf.q[1] = *(const uint4*)(bp + 16);
        acc = __builtin_amdgcn_wmma_f32_16x16x32_bf16(
            false, af.v, false, bf.v, (short)0, acc, false, false);
    }

    const int col = tn * 16 + lr;
    const float bv = bias ? bias[col] : 0.0f;
    #pragma unroll
    for (int j = 0; j < 8; ++j) {
        const int row = tm * 16 + khalf * 8 + j;
        const float val = (acc[j] + bv) * outScale;
        const long long idx = (long long)row * ldo + col;
        if (outF) outF[idx] = val;
        else      outB[idx] = f2bf(val);
    }
}

// ---------------- SR conv as implicit GEMM (no patch buffer) ----------------
// xsr[b*NSR+psr][co] = sum_{pix,ci} xb[b, tok(psr,pix), ci] * WsrP[co][pix*128+ci] + bsr[co]
__global__ __launch_bounds__(256) void conv_gemm_wmma(
    const unsigned short* __restrict__ xb,      // [B, NTOK, CDIM] bf16
    const unsigned short* __restrict__ WsrP,    // [CDIM, KPATCH] bf16, K-order (kh,kw,ci)
    float* __restrict__ outF,                   // [B*NSR, CDIM]
    const float* __restrict__ bias)
{
    const int tilesN = CDIM >> 4;               // 8
    const int wave = blockIdx.x * (blockDim.x >> 5) + (threadIdx.x >> 5);
    if (wave >= (BATCH * NSR / 16) * tilesN) return;
    const int tm = wave / tilesN;
    const int tn = wave % tilesN;

    const int lane  = threadIdx.x & 31;
    const int lr    = lane & 15;
    const int khalf = lane >> 4;

    const int row = tm * 16 + lr;               // b*NSR + psr
    const int b   = row >> 8;
    const int psr = row & 255;
    const int ph  = psr >> 4;
    const int pw  = psr & 15;

    const unsigned short* bRow = WsrP + (long long)(tn * 16 + lr) * KPATCH;

    v8f acc = {};
    for (int k = 0; k < KPATCH; k += 32) {
        const int pix = k >> 7;                 // fixed within a 32-chunk
        const int kh  = pix >> 2;
        const int kw  = pix & 3;
        const int ci0 = (k & 127) + khalf * 8;
        const int tok = (ph * SRF + kh) * WW + (pw * SRF + kw);
        union { v16bf v; uint4 q[2]; } af, bf;
        const unsigned short* ap = xb + ((long long)(b * NTOK + tok) * CDIM + ci0);
        af.q[0] = *(const uint4*)(ap);
        af.q[1] = *(const uint4*)(ap + 16);
        const unsigned short* bp = bRow + k + khalf * 8;
        bf.q[0] = *(const uint4*)(bp);
        bf.q[1] = *(const uint4*)(bp + 16);
        acc = __builtin_amdgcn_wmma_f32_16x16x32_bf16(
            false, af.v, false, bf.v, (short)0, acc, false, false);
    }

    const int col = tn * 16 + lr;
    const float bv = bias[col];
    #pragma unroll
    for (int j = 0; j < 8; ++j) {
        const int orow = tm * 16 + khalf * 8 + j;
        outF[(long long)orow * CDIM + col] = acc[j] + bv;
    }
}

// ---------------- channel LayerNorm: f32 [rows,128] -> bf16 ----------------
__global__ __launch_bounds__(256) void layernorm_rows(const float* __restrict__ x,
                                                      const float* __restrict__ gamma,
                                                      const float* __restrict__ beta,
                                                      unsigned short* __restrict__ out,
                                                      int rows) {
    const int row = blockIdx.x * (blockDim.x >> 5) + (threadIdx.x >> 5);
    if (row >= rows) return;
    const int lane = threadIdx.x & 31;
    float4 v = *(const float4*)(x + (long long)row * CDIM + lane * 4);
    float s = v.x + v.y + v.z + v.w;
    #pragma unroll
    for (int m = 16; m >= 1; m >>= 1) s += __shfl_xor(s, m, 32);
    const float mu = s * (1.0f / CDIM);
    float4 d = make_float4(v.x - mu, v.y - mu, v.z - mu, v.w - mu);
    float vs = d.x * d.x + d.y * d.y + d.z * d.z + d.w * d.w;
    #pragma unroll
    for (int m = 16; m >= 1; m >>= 1) vs += __shfl_xor(vs, m, 32);
    const float rs = rsqrtf(vs * (1.0f / CDIM) + 1e-5f);
    const int c = lane * 4;
    ushort4 o;
    o.x = f2bf(d.x * rs * gamma[c + 0] + beta[c + 0]);
    o.y = f2bf(d.y * rs * gamma[c + 1] + beta[c + 1]);
    o.z = f2bf(d.z * rs * gamma[c + 2] + beta[c + 2]);
    o.w = f2bf(d.w * rs * gamma[c + 3] + beta[c + 3]);
    *(ushort4*)(out + (long long)row * CDIM + c) = o;
}

// ---------------- repack kv: f32 [B*NSR, 2C] -> k bf16 [B*H, NSR, DH], v^T bf16 [B*H, DH, NSR] ----------------
__global__ __launch_bounds__(256) void repack_kv(const float* __restrict__ kv,
                                                 unsigned short* __restrict__ kbuf,
                                                 unsigned short* __restrict__ vt) {
    long long i = (long long)blockIdx.x * blockDim.x + threadIdx.x;
    const long long total = (long long)BATCH * NSR * 2 * CDIM;
    if (i >= total) return;
    const int c   = (int)(i % (2 * CDIM));
    const int row = (int)(i / (2 * CDIM));
    const int nsr = row % NSR;
    const int b   = row / NSR;
    const unsigned short bv = f2bf(kv[i]);
    if (c < CDIM) {
        const int hh = c >> 6, d = c & 63;
        kbuf[(((long long)(b * HEADS + hh) * NSR) + nsr) * DH + d] = bv;
    } else {
        const int c2 = c - CDIM;
        const int hh = c2 >> 6, d = c2 & 63;
        vt[(((long long)(b * HEADS + hh) * DH) + d) * NSR + nsr] = bv;
    }
}

// ---------------- fused attention: scores -> softmax -> PV, all on-chip ----------------
// grid: (NTOK/128, 1, B*HEADS); 8 waves/block, one 16-row q-tile per wave.
// q is pre-scaled bf16. K,V read from global (L2-resident, ~1MB each).
#define PSTRIDE 40   // padded row stride (ushorts) for conflict-free ds_load_b128
__global__ __launch_bounds__(256) void attn_fused(
    const unsigned short* __restrict__ qb,      // [B, NTOK, CDIM]
    const unsigned short* __restrict__ kbuf,    // [B*H, NSR, DH]
    const unsigned short* __restrict__ vt,      // [B*H, DH, NSR]
    unsigned short* __restrict__ attnout)       // [B, NTOK, CDIM]
{
    __shared__ __align__(16) unsigned short pstage[8][16 * PSTRIDE];

    const int bz = blockIdx.z;
    const int b  = bz / HEADS;
    const int h  = bz % HEADS;
    const int wv = threadIdx.x >> 5;
    const int lane  = threadIdx.x & 31;
    const int lr    = lane & 15;
    const int khalf = lane >> 4;
    const int row0  = (blockIdx.x * 8 + wv) * 16;

    union U { v16bf v; uint4 q[2]; };

    // q A-fragments (K=64 -> 2 chunks), kept in registers
    const unsigned short* qrow = qb + ((long long)(b * NTOK + row0 + lr) * CDIM) + h * DH;
    U aq[2];
    #pragma unroll
    for (int c = 0; c < 2; ++c) {
        const unsigned short* ap = qrow + c * 32 + khalf * 8;
        aq[c].q[0] = *(const uint4*)(ap);
        aq[c].q[1] = *(const uint4*)(ap + 16);
    }

    // scores: 16 C-tiles of 16x16 held in registers (128 VGPRs)
    const unsigned short* kb = kbuf + (long long)bz * NSR * DH;
    v8f sc[16];
    #pragma unroll
    for (int tn = 0; tn < 16; ++tn) {
        v8f a = {};
        const unsigned short* brow = kb + (tn * 16 + lr) * DH;
        #pragma unroll
        for (int c = 0; c < 2; ++c) {
            U bfr;
            const unsigned short* bp = brow + c * 32 + khalf * 8;
            bfr.q[0] = *(const uint4*)(bp);
            bfr.q[1] = *(const uint4*)(bp + 16);
            a = __builtin_amdgcn_wmma_f32_16x16x32_bf16(
                false, aq[c].v, false, bfr.v, (short)0, a, false, false);
        }
        sc[tn] = a;
    }

    // softmax over 256 columns per row. C-layout: row j in VGPR j (lanes 0-15),
    // row 8+j in VGPR j (lanes 16-31) -> reduce within 16-lane half with masks 8..1.
    float inv[8];
    #pragma unroll
    for (int j = 0; j < 8; ++j) {
        float m = sc[0][j];
        #pragma unroll
        for (int tn = 1; tn < 16; ++tn) m = fmaxf(m, sc[tn][j]);
        #pragma unroll
        for (int msk = 8; msk >= 1; msk >>= 1) m = fmaxf(m, __shfl_xor(m, msk, 32));
        float s = 0.f;
        #pragma unroll
        for (int tn = 0; tn < 16; ++tn) {
            float e = __expf(sc[tn][j] - m);
            sc[tn][j] = e;
            s += e;
        }
        #pragma unroll
        for (int msk = 8; msk >= 1; msk >>= 1) s += __shfl_xor(s, msk, 32);
        inv[j] = 1.0f / s;
    }

    // PV: out[16,64] = P[16,256] x V[256,64]; stage P 32 columns at a time through LDS
    // to convert C-fragment layout -> A-fragment layout (per-wave region, in-order LDS).
    unsigned short* st = &pstage[wv][0];
    const unsigned short* vb = vt + (long long)bz * DH * NSR;
    v8f oacc0 = {}, oacc1 = {}, oacc2 = {}, oacc3 = {};
    #pragma unroll
    for (int kk = 0; kk < 8; ++kk) {
        #pragma unroll
        for (int t = 0; t < 2; ++t) {
            const int tn = kk * 2 + t;
            #pragma unroll
            for (int j = 0; j < 8; ++j) {
                const int prow = khalf * 8 + j;
                st[prow * PSTRIDE + t * 16 + lr] = f2bf(sc[tn][j] * inv[j]);
            }
        }
        asm volatile("" ::: "memory");   // keep ds_load after the staging stores
        U pa;
        const unsigned short* lp = st + lr * PSTRIDE + khalf * 8;
        pa.q[0] = *(const uint4*)(lp);
        pa.q[1] = *(const uint4*)(lp + 16);
        asm volatile("" ::: "memory");   // keep next iter's stores after this load
        #pragma unroll
        for (int tnn = 0; tnn < 4; ++tnn) {
            U bfr;
            const unsigned short* bp = vb + (tnn * 16 + lr) * NSR + kk * 32 + khalf * 8;
            bfr.q[0] = *(const uint4*)(bp);
            bfr.q[1] = *(const uint4*)(bp + 16);
            v8f* oa = (tnn == 0) ? &oacc0 : (tnn == 1) ? &oacc1 : (tnn == 2) ? &oacc2 : &oacc3;
            *oa = __builtin_amdgcn_wmma_f32_16x16x32_bf16(
                false, pa.v, false, bfr.v, (short)0, *oa, false, false);
        }
    }

    // store merged-head output (bf16)
    v8f oaccs[4] = { oacc0, oacc1, oacc2, oacc3 };
    #pragma unroll
    for (int tnn = 0; tnn < 4; ++tnn) {
        const int d = tnn * 16 + lr;
        #pragma unroll
        for (int j = 0; j < 8; ++j) {
            const int r = row0 + khalf * 8 + j;
            attnout[((long long)(b * NTOK + r) * CDIM) + h * DH + d] = f2bf(oaccs[tnn][j]);
        }
    }
}

// ---------------- host-side orchestration ----------------
static inline size_t alignUp(size_t v) { return (v + 255) & ~(size_t)255; }

extern "C" void kernel_launch(void* const* d_in, const int* in_sizes, int n_in,
                              void* d_out, int out_size, void* d_ws, size_t ws_size,
                              hipStream_t stream) {
    const float* x     = (const float*)d_in[0];
    const float* Wq    = (const float*)d_in[1];
    const float* bq    = (const float*)d_in[2];
    const float* Wkv   = (const float*)d_in[3];
    const float* bkv   = (const float*)d_in[4];
    const float* Wsr   = (const float*)d_in[5];
    const float* bsr   = (const float*)d_in[6];
    const float* gamma = (const float*)d_in[7];
    const float* beta  = (const float*)d_in[8];
    const float* Wproj = (const float*)d_in[9];
    const float* bproj = (const float*)d_in[10];
    float* out = (float*)d_out;

    char* ws = (char*)d_ws;
    size_t off = 0;
    auto take = [&](size_t bytes) { char* p = ws + off; off = alignUp(off + bytes); return p; };

    unsigned short* xb      = (unsigned short*)take((size_t)BATCH * NTOK * CDIM * 2);        // 16.8 MB
    unsigned short* qb      = (unsigned short*)take((size_t)BATCH * NTOK * CDIM * 2);        // 16.8 MB
    unsigned short* attnout = (unsigned short*)take((size_t)BATCH * NTOK * CDIM * 2);        // 16.8 MB
    float*          xsrraw  = (float*)take((size_t)BATCH * NSR * CDIM * 4);                  // 2.1 MB
    float*          kvraw   = (float*)take((size_t)BATCH * NSR * 2 * CDIM * 4);              // 4.2 MB
    unsigned short* xsrln   = (unsigned short*)take((size_t)BATCH * NSR * CDIM * 2);         // 1 MB
    unsigned short* kbuf    = (unsigned short*)take((size_t)BATCH * HEADS * NSR * DH * 2);   // 1 MB
    unsigned short* vt      = (unsigned short*)take((size_t)BATCH * HEADS * DH * NSR * 2);   // 1 MB
    unsigned short* Wqb     = (unsigned short*)take((size_t)CDIM * CDIM * 2);
    unsigned short* Wkvb    = (unsigned short*)take((size_t)2 * CDIM * CDIM * 2);
    unsigned short* Wsrb    = (unsigned short*)take((size_t)CDIM * KPATCH * 2);
    unsigned short* Wprojb  = (unsigned short*)take((size_t)CDIM * CDIM * 2);
    (void)ws_size; (void)in_sizes; (void)n_in; (void)out_size;

    // 1) fp32 -> bf16 conversions (+ Wsr K-order permutation)
    {
        int n = BATCH * NTOK * CDIM;
        cvt_f32_bf16<<<(n + 255) / 256, 256, 0, stream>>>(x, xb, n);
        cvt_f32_bf16<<<64, 256, 0, stream>>>(Wq, Wqb, CDIM * CDIM);
        cvt_f32_bf16<<<128, 256, 0, stream>>>(Wkv, Wkvb, 2 * CDIM * CDIM);
        cvt_wsr_perm<<<1024, 256, 0, stream>>>(Wsr, Wsrb);
        cvt_f32_bf16<<<64, 256, 0, stream>>>(Wproj, Wprojb, CDIM * CDIM);
    }
    // 2) q = (x @ Wq^T + bq) * dh^-0.5  -> bf16
    {
        int tiles = (BATCH * NTOK / 16) * (CDIM / 16);
        gemm_bf16_wmma<<<(tiles + 7) / 8, 256, 0, stream>>>(
            xb, CDIM, Wqb, CDIM, nullptr, qb, CDIM, bq,
            BATCH * NTOK, CDIM, CDIM, 0.125f);
    }
    // 3) SR conv as implicit GEMM -> f32
    {
        int tiles = (BATCH * NSR / 16) * (CDIM / 16);
        conv_gemm_wmma<<<(tiles + 7) / 8, 256, 0, stream>>>(xb, Wsrb, xsrraw, bsr);
    }
    // 4) LayerNorm over channels -> bf16
    layernorm_rows<<<(BATCH * NSR + 7) / 8, 256, 0, stream>>>(xsrraw, gamma, beta, xsrln, BATCH * NSR);
    // 5) kv = xsr_ln @ Wkv^T + bkv -> f32
    {
        int tiles = (BATCH * NSR / 16) * (2 * CDIM / 16);
        gemm_bf16_wmma<<<(tiles + 7) / 8, 256, 0, stream>>>(
            xsrln, CDIM, Wkvb, CDIM, kvraw, nullptr, 2 * CDIM, bkv,
            BATCH * NSR, 2 * CDIM, CDIM, 1.0f);
    }
    // 6) split heads: k row-major, v transposed, bf16
    {
        long long total = (long long)BATCH * NSR * 2 * CDIM;
        repack_kv<<<(int)((total + 255) / 256), 256, 0, stream>>>(kvraw, kbuf, vt);
    }
    // 7) fused attention: QK^T + softmax + PV, scores never touch HBM
    attn_fused<<<dim3(NTOK / 128, 1, BATCH * HEADS), 256, 0, stream>>>(qb, kbuf, vt, attnout);
    // 8) out = attnout @ Wproj^T + bproj -> f32 (final output)
    {
        int tiles = (BATCH * NTOK / 16) * (CDIM / 16);
        gemm_bf16_wmma<<<(tiles + 7) / 8, 256, 0, stream>>>(
            attnout, CDIM, Wprojb, CDIM, out, nullptr, CDIM, bproj,
            BATCH * NTOK, CDIM, CDIM, 1.0f);
    }
}
